// ClipLoss_2190433321148
// MI455X (gfx1250) — compile-verified, matrix operand order checked by reading, not run on hardware
//
#include <hip/hip_runtime.h>
#include <hip/hip_bf16.h>
#include <math.h>

// ---------------------------------------------------------------------------
// CLIP contrastive loss for MI455X (gfx1250, wave32, WMMA).
//
//   loss = 0.5*mean(lse_rows(s*I@T^T) - diag) + 0.5*mean(lse_rows(s*T@I^T) - diag)
//
// Pipeline:
//   1) split_f32_to_f16hilo : x -> (hi, lo) f16 pair (Markidis split), both mats
//   2) lse_rows_kernel(I,T) -> lse_i ; lse_rows_kernel(T,I) -> lse_t
//      fused GEMM (3-term f16 WMMA) + streaming online logsumexp per row;
//      the 1 GiB logits matrix never touches memory.
//      Register blocking: 32x32 per-wave tile (2x2 of 16x16) -> each fragment
//      feeds 2 WMMA-pairs; per K-step only 4 global + 4 LDS b128-frags per
//      12 WMMAs (2x less WGP$ fill than a 16x64 tiling).
//   3) diag_kernel  : diag[i] = s * dot(I[i], T[i])  (fp32, exact path)
//   4) final_reduce : scalar loss
//
// Workspace: 4 * N*D f16 arrays (64 MiB) + 3 * N f32 arrays (192 KiB).
// ---------------------------------------------------------------------------

typedef __attribute__((ext_vector_type(16))) _Float16 v16h;
typedef __attribute__((ext_vector_type(8)))  float    v8f;

union Frag16 { v16h v; uint4 q[2]; };
union H4     { _Float16 h[4]; uint2 u; };

#define WMMA_F16(A, B, C) \
  __builtin_amdgcn_wmma_f32_16x16x32_f16(false, (A), false, (B), (short)0, (C), false, false)

// ---------------------------------------------------------------------------
// 1) fp32 -> (hi, lo) f16 split
// ---------------------------------------------------------------------------
__global__ __launch_bounds__(256) void split_f32_to_f16hilo(
    const float* __restrict__ src, _Float16* __restrict__ hi,
    _Float16* __restrict__ lo, int n4)
{
  int idx = blockIdx.x * 256 + threadIdx.x;
  if (idx >= n4) return;
  float4 x = ((const float4*)src)[idx];
  float v[4] = {x.x, x.y, x.z, x.w};
  H4 hh, ll;
#pragma unroll
  for (int i = 0; i < 4; ++i) {
    _Float16 h = (_Float16)v[i];
    hh.h[i] = h;
    ll.h[i] = (_Float16)(v[i] - (float)h);
  }
  ((uint2*)hi)[idx] = hh.u;
  ((uint2*)lo)[idx] = ll.u;
}

// ---------------------------------------------------------------------------
// 2) Fused GEMM + streaming row logsumexp.
//    Block: 256 threads = 8 waves, owns 128 rows of A (full K=512, hi+lo in
//    256 KiB LDS). Waves pair up: strip = wave/2 owns rows [strip*32,+32),
//    phase = wave&1 sweeps alternate 32-column groups. Per column group each
//    wave computes a 32x32 tile = 2x2 WMMA tiles with full K accumulation,
//    then updates per-row online (max,sum). Phase pairs merge stats via LDS.
//    Fragment layout = ISA 16-bit per-lane K interleave:
//      lane (lh = lane&15 row, half = lane>>4), per-lane K order:
//      [kb+half*8 .. +7] ++ [kb+16+half*8 .. +7]  (two 16B chunks).
// ---------------------------------------------------------------------------
__global__ __launch_bounds__(256) void lse_rows_kernel(
    const _Float16* __restrict__ Ahi, const _Float16* __restrict__ Alo,
    const _Float16* __restrict__ Bhi, const _Float16* __restrict__ Blo,
    const float* __restrict__ scale_p, float* __restrict__ lse_out, int N)
{
  extern __shared__ char smem_raw[];
  _Float16* sAhi = (_Float16*)smem_raw;                 // [128*512]
  _Float16* sAlo = sAhi + 128 * 512;                    // [128*512]
  float* mbuf = (float*)(smem_raw + 2 * 128 * 512 * sizeof(_Float16)); // [2][128]
  float* sbuf = mbuf + 256;                                            // [2][128]

  const int r0 = blockIdx.x * 128;

  // cooperative stage of A hi/lo rows into LDS (uint4 = 8 halves per copy)
  {
    const uint4* gH = (const uint4*)(Ahi + (size_t)r0 * 512);
    const uint4* gL = (const uint4*)(Alo + (size_t)r0 * 512);
    uint4* sH = (uint4*)sAhi;
    uint4* sL = (uint4*)sAlo;
    for (int i = threadIdx.x; i < 128 * 512 / 8; i += 256) {
      sH[i] = gH[i];
      sL[i] = gL[i];
    }
  }
  __syncthreads();

  const int wave  = threadIdx.x >> 5;
  const int lane  = threadIdx.x & 31;
  const int lh    = lane & 15;     // row (A) / column (B) within a 16-tile
  const int half  = lane >> 4;     // K-half selector
  const int strip = wave >> 1;     // 0..3 : which 32-row strip
  const int phase = wave & 1;      // 0/1  : which half of the column groups

  const _Float16* aH[2];
  const _Float16* aL[2];
  aH[0] = sAhi + (strip * 32 + lh) * 512 + half * 8;
  aH[1] = aH[0] + 16 * 512;
  aL[0] = sAlo + (strip * 32 + lh) * 512 + half * 8;
  aL[1] = aL[0] + 16 * 512;

  const float sc = *scale_p;

  float m[2][8], s[2][8];
#pragma unroll
  for (int rt = 0; rt < 2; ++rt)
#pragma unroll
    for (int j = 0; j < 8; ++j) { m[rt][j] = -INFINITY; s[rt][j] = 0.0f; }

  const int ncg = N >> 5;                          // 512 groups of 32 columns
  for (int cg = phase; cg < ncg; cg += 2) {
    const int n0 = cg << 5;

    const _Float16* bHp[2];
    const _Float16* bLp[2];
#pragma unroll
    for (int ct = 0; ct < 2; ++ct) {
      size_t roff = (size_t)(n0 + ct * 16 + lh) * 512 + half * 8;
      bHp[ct] = Bhi + roff;
      bLp[ct] = Blo + roff;
    }

    // prefetch this wave's next column group (64 columns ahead) into cache
    if (cg + 2 < ncg) {
      __builtin_prefetch(bHp[0] + 64 * 512, 0, 3);
      __builtin_prefetch(bLp[0] + 64 * 512, 0, 3);
      __builtin_prefetch(bHp[1] + 64 * 512, 0, 3);
      __builtin_prefetch(bLp[1] + 64 * 512, 0, 3);
    }

    v8f acc[2][2];
#pragma unroll
    for (int rt = 0; rt < 2; ++rt)
#pragma unroll
      for (int ct = 0; ct < 2; ++ct)
#pragma unroll
        for (int j = 0; j < 8; ++j) acc[rt][ct][j] = 0.0f;

#pragma unroll 2
    for (int kk = 0; kk < 16; ++kk) {
      const int kb = kk << 5;                      // K base (step 32)
      Frag16 ah[2], al[2], bh[2], bl[2];
#pragma unroll
      for (int rt = 0; rt < 2; ++rt) {
        ah[rt].q[0] = *(const uint4*)(aH[rt] + kb);
        ah[rt].q[1] = *(const uint4*)(aH[rt] + kb + 16);
        al[rt].q[0] = *(const uint4*)(aL[rt] + kb);
        al[rt].q[1] = *(const uint4*)(aL[rt] + kb + 16);
      }
#pragma unroll
      for (int ct = 0; ct < 2; ++ct) {
        bh[ct].q[0] = *(const uint4*)(bHp[ct] + kb);
        bh[ct].q[1] = *(const uint4*)(bHp[ct] + kb + 16);
        bl[ct].q[0] = *(const uint4*)(bLp[ct] + kb);
        bl[ct].q[1] = *(const uint4*)(bLp[ct] + kb + 16);
      }
      // 3-term split-f16 product: hi*hi + hi*lo + lo*hi  (~2^-22 rel err)
#pragma unroll
      for (int rt = 0; rt < 2; ++rt)
#pragma unroll
        for (int ct = 0; ct < 2; ++ct) {
          acc[rt][ct] = WMMA_F16(ah[rt].v, bh[ct].v, acc[rt][ct]);
          acc[rt][ct] = WMMA_F16(ah[rt].v, bl[ct].v, acc[rt][ct]);
          acc[rt][ct] = WMMA_F16(al[rt].v, bh[ct].v, acc[rt][ct]);
        }
    }

    // online logsumexp update: each lane owns one column slot of each tile,
    // VGPR j of tile rt holds row (rt*16 + half*8 + j) of the 32-row strip.
#pragma unroll
    for (int rt = 0; rt < 2; ++rt)
#pragma unroll
      for (int ct = 0; ct < 2; ++ct)
#pragma unroll
        for (int j = 0; j < 8; ++j) {
          float v  = sc * acc[rt][ct][j];
          float nm = fmaxf(m[rt][j], v);
          s[rt][j] = s[rt][j] * __expf(m[rt][j] - nm) + __expf(v - nm);
          m[rt][j] = nm;
        }
  }

  // cross-lane LSE merge within each 16-lane half (cols 0..15 of the tile)
#pragma unroll
  for (int off = 1; off < 16; off <<= 1) {
#pragma unroll
    for (int rt = 0; rt < 2; ++rt)
#pragma unroll
      for (int j = 0; j < 8; ++j) {
        float om = __shfl_xor(m[rt][j], off, 32);
        float os = __shfl_xor(s[rt][j], off, 32);
        float nm = fmaxf(m[rt][j], om);
        s[rt][j] = s[rt][j] * __expf(m[rt][j] - nm) + os * __expf(om - nm);
        m[rt][j] = nm;
      }
  }

  // publish per-phase stats, then merge phase pairs and write lse
  if (lh == 0) {                                   // lanes 0 and 16 of each wave
#pragma unroll
    for (int rt = 0; rt < 2; ++rt)
#pragma unroll
      for (int j = 0; j < 8; ++j) {
        int row = strip * 32 + rt * 16 + half * 8 + j;   // 0..127
        mbuf[phase * 128 + row] = m[rt][j];
        sbuf[phase * 128 + row] = s[rt][j];
      }
  }
  __syncthreads();
  if (threadIdx.x < 128) {
    int row = threadIdx.x;
    float m0 = mbuf[row],       m1 = mbuf[128 + row];
    float s0 = sbuf[row],       s1 = sbuf[128 + row];
    float nm = fmaxf(m0, m1);
    lse_out[r0 + row] = nm + __logf(s0 * __expf(m0 - nm) + s1 * __expf(m1 - nm));
  }
}

// ---------------------------------------------------------------------------
// 3) diag[i] = scale * dot(I[i], T[i])  — one wave per row, fp32-exact.
// ---------------------------------------------------------------------------
__global__ __launch_bounds__(256) void diag_kernel(
    const float* __restrict__ I, const float* __restrict__ T,
    const float* __restrict__ scale_p, float* __restrict__ diag, int N)
{
  const int wave = threadIdx.x >> 5;
  const int lane = threadIdx.x & 31;
  const int row  = blockIdx.x * 8 + wave;
  if (row >= N) return;

  const float4* a = (const float4*)(I + (size_t)row * 512);
  const float4* b = (const float4*)(T + (size_t)row * 512);
  float sum = 0.0f;
#pragma unroll
  for (int i = 0; i < 4; ++i) {                    // 512/4 = 128 float4, 4/lane
    float4 xa = a[i * 32 + lane];
    float4 xb = b[i * 32 + lane];
    sum += xa.x * xb.x + xa.y * xb.y + xa.z * xb.z + xa.w * xb.w;
  }
#pragma unroll
  for (int off = 16; off > 0; off >>= 1)
    sum += __shfl_xor(sum, off, 32);
  if (lane == 0) diag[row] = (*scale_p) * sum;
}

// ---------------------------------------------------------------------------
// 4) loss = sum(lse_i + lse_t - 2*diag) / (2N)
// ---------------------------------------------------------------------------
__global__ __launch_bounds__(256) void final_reduce_kernel(
    const float* __restrict__ lse_i, const float* __restrict__ lse_t,
    const float* __restrict__ diag, float* __restrict__ out, int N)
{
  __shared__ float red[256];
  float acc = 0.0f;
  for (int i = threadIdx.x; i < N; i += 256)
    acc += lse_i[i] + lse_t[i] - 2.0f * diag[i];
  red[threadIdx.x] = acc;
  __syncthreads();
#pragma unroll
  for (int off = 128; off > 0; off >>= 1) {
    if (threadIdx.x < off) red[threadIdx.x] += red[threadIdx.x + off];
    __syncthreads();
  }
  if (threadIdx.x == 0) out[0] = red[0] / (2.0f * (float)N);
}

// ---------------------------------------------------------------------------
extern "C" void kernel_launch(void* const* d_in, const int* in_sizes, int n_in,
                              void* d_out, int out_size, void* d_ws, size_t ws_size,
                              hipStream_t stream)
{
  const float* img = (const float*)d_in[0];
  const float* txt = (const float*)d_in[1];
  const float* scl = (const float*)d_in[2];

  const int D = 512;
  const int N = in_sizes[0] / D;                   // 16384

  char* ws = (char*)d_ws;
  const size_t half_bytes = (size_t)N * D * sizeof(_Float16);  // 16 MiB
  _Float16* Ihi = (_Float16*)(ws + 0 * half_bytes);
  _Float16* Ilo = (_Float16*)(ws + 1 * half_bytes);
  _Float16* Thi = (_Float16*)(ws + 2 * half_bytes);
  _Float16* Tlo = (_Float16*)(ws + 3 * half_bytes);
  float* lse_i  = (float*)(ws + 4 * half_bytes);
  float* lse_t  = lse_i + N;
  float* diag   = lse_t + N;

  const int n4 = N * D / 4;
  split_f32_to_f16hilo<<<(n4 + 255) / 256, 256, 0, stream>>>(img, Ihi, Ilo, n4);
  split_f32_to_f16hilo<<<(n4 + 255) / 256, 256, 0, stream>>>(txt, Thi, Tlo, n4);

  // 256 KiB A-tile (hi+lo) + 2 KiB stat-merge region
  const size_t lds_bytes = (size_t)2 * 128 * 512 * sizeof(_Float16) + 2048;
  lse_rows_kernel<<<N / 128, 256, lds_bytes, stream>>>(Ihi, Ilo, Thi, Tlo, scl, lse_i, N);
  lse_rows_kernel<<<N / 128, 256, lds_bytes, stream>>>(Thi, Tlo, Ihi, Ilo, scl, lse_t, N);

  diag_kernel<<<N / 8, 256, 0, stream>>>(img, txt, scl, diag, N);
  final_reduce_kernel<<<1, 256, 0, stream>>>(lse_i, lse_t, diag, (float*)d_out, N);
}
